// MPGNN_78340203479710
// MI455X (gfx1250) — matrix operand order repaired and checked by reading, not compiled, not run on hardware
//
#include <hip/hip_runtime.h>
#include <hip/hip_bf16.h>

typedef __attribute__((ext_vector_type(16))) _Float16 v16h;
typedef __attribute__((ext_vector_type(8)))  float    v8f;
typedef __attribute__((ext_vector_type(4)))  unsigned int u32x4;

static constexpr int Rres  = 128;
static constexpr int Vn    = 6 * 32 * 32;  // 6144 nodes
static constexpr int En    = 4 * Vn;       // 24576 edges

// ---------------------------------------------------------------------------
// Generic WMMA GEMM:  C[M,N] = A[M,K](f32) * B[K,N](f32) + bias (ReLU opt.)
// with compile-time-selected output row remap and f16/f32 output.
// Requires M%128==0, N%32==0, K%32==0 (guaranteed by all call sites).
// Block: 256 threads = 8 waves, tile 128(M) x 32(N), K-step 32.
// ---------------------------------------------------------------------------
#define GBM 128
#define GBN 32
#define GKC 32
#define APAD 40   // row pad: 40 f16 = 80B, keeps 16B alignment, spreads banks

template<bool RELU, bool HALFOUT, bool RMAP>
__global__ __launch_bounds__(256)
void gemm_wmma(const float* __restrict__ A, int lda,
               const float* __restrict__ B, int ldb,
               void* __restrict__ Cv, int ldc,
               const float* __restrict__ bias,
               const int* __restrict__ row_map,
               int M, int N, int K)
{
  __shared__ alignas(16) _Float16 As[GBM][APAD];
  __shared__ alignas(16) _Float16 Bs[GBN][APAD];

  const int tid  = threadIdx.x;
  const int lane = tid & 31;
  const int w    = tid >> 5;
  const int wm   = w >> 1;          // 0..3 : 32-row strip
  const int wn   = w & 1;           // 0..1 : 16-col strip
  const int bm   = blockIdx.x;
  const int bn   = blockIdx.y;

  const int l15 = lane & 15;
  const int hi  = lane >> 4;        // 0/1 (lane half)
  const int aq0 = hi;               // A frag: 16B chunk index (khalf/8)
  const int bq0 = hi;               // B frag: 16B chunk index (khalf/8)

  v8f acc[2] = {};

  for (int k0 = 0; k0 < K; k0 += GKC) {
    // stage A tile (128x32 f32 -> f16), 16 elems / thread
    {
      const int r  = tid >> 1;
      const int c0 = (tid & 1) * 16;
      const float* ga = A + (size_t)(bm * GBM + r) * lda + k0 + c0;
      _Float16* sa = &As[r][c0];
      #pragma unroll
      for (int i = 0; i < 16; ++i) sa[i] = (_Float16)ga[i];
    }
    // stage B tile transposed (32x32): Bs[n][k], 4 elems / thread
    {
      const int kk = tid >> 3;
      const int n0 = (tid & 7) * 4;
      const float* gb = B + (size_t)(k0 + kk) * ldb + bn * GBN + n0;
      #pragma unroll
      for (int i = 0; i < 4; ++i) Bs[n0 + i][kk] = (_Float16)gb[i];
    }
    __syncthreads();

    // B fragment: col = lane&15, K chunks {khalf..+7, 16+khalf..+7}
    union { v16h v; u32x4 q[2]; } bf;
    {
      const u32x4* brow = (const u32x4*)&Bs[wn * 16 + l15][0];
      bf.q[0] = brow[bq0];
      bf.q[1] = brow[bq0 + 2];
    }
    #pragma unroll
    for (int s = 0; s < 2; ++s) {
      // A fragment: row = lane&15, K chunks {khalf..+7, 16+khalf..+7}
      union { v16h v; u32x4 q[2]; } af;
      const u32x4* arow = (const u32x4*)&As[wm * 32 + s * 16 + l15][0];
      af.q[0] = arow[aq0];
      af.q[1] = arow[2 + aq0];
      acc[s] = __builtin_amdgcn_wmma_f32_16x16x32_f16(
          false, af.v, false, bf.v, (short)0, acc[s], false, false);
    }
    __syncthreads();
  }

  // epilogue: C/D layout — vgpr p, row = p + (lane>=16?8:0), col = lane&15
  const int gc = bn * GBN + wn * 16 + l15;
  const float bv = bias[gc];
  #pragma unroll
  for (int s = 0; s < 2; ++s) {
    const int grBase = bm * GBM + wm * 32 + s * 16 + hi * 8;
    #pragma unroll
    for (int p = 0; p < 8; ++p) {
      const int gr = grBase + p;
      const int ro = RMAP ? row_map[gr] : gr;
      float vv = acc[s][p] + bv;
      if (RELU) vv = vv > 0.0f ? vv : 0.0f;
      if (HALFOUT) ((_Float16*)Cv)[(size_t)ro * ldc + gc] = (_Float16)vv;
      else         ((float*)Cv)[(size_t)ro * ldc + gc]    = vv;
    }
  }
}

// ---------------------------------------------------------------------------
// Encoder: 4x4 mean pool per face: (6*128*128, 64) -> (6144, 64)
// ---------------------------------------------------------------------------
__global__ __launch_bounds__(256)
void pool4x4_kernel(const float* __restrict__ in, float* __restrict__ out)
{
  int t = blockIdx.x * blockDim.x + threadIdx.x;
  if (t >= Vn * 64) return;
  int c = t & 63, v = t >> 6;
  int f = v >> 10, rem = v & 1023, i = rem >> 5, j = rem & 31;
  float s = 0.0f;
  for (int a = 0; a < 4; ++a)
    for (int b = 0; b < 4; ++b) {
      int pix = (f * Rres + i * 4 + a) * Rres + j * 4 + b;
      s += in[(size_t)pix * 64 + c];
    }
  out[t] = s * 0.0625f;
}

// agg[v][o] = conv_b[o]
__global__ __launch_bounds__(256)
void agg_init_kernel(float* __restrict__ agg, const float* __restrict__ cb)
{
  int t = blockIdx.x * blockDim.x + threadIdx.x;
  if (t < Vn * 64) agg[t] = cb[t & 63];
}

__global__ __launch_bounds__(256)
void relu_kernel(float* __restrict__ x, int n)
{
  int t = blockIdx.x * blockDim.x + threadIdx.x;
  if (t < n) x[t] = fmaxf(x[t], 0.0f);
}

// ---------------------------------------------------------------------------
// NNConv message + scatter: one wave per edge.
// m[e,o] = sum_h nf[src[e],h] * we16[e, h*64+o];  atomicAdd(agg[dst[e],o])
// Lane l owns outputs o=2l, 2l+1 (one dword of the f16 row per h).
// ---------------------------------------------------------------------------
__global__ __launch_bounds__(256)
void nnconv_msg_kernel(const float* __restrict__ nf,
                       const _Float16* __restrict__ we16,
                       const int* __restrict__ src,
                       const int* __restrict__ dst,
                       float* __restrict__ agg)
{
  int gw   = (blockIdx.x * blockDim.x + threadIdx.x) >> 5;  // edge id
  int lane = threadIdx.x & 31;
  if (gw >= En) return;

  const int s = src[gw];
  const int d = dst[gw];
  const float a0 = nf[(size_t)s * 64 + lane];
  const float a1 = nf[(size_t)s * 64 + 32 + lane];
  const _Float16* wrow = we16 + (size_t)gw * 4096;

  // speculative prefetch of the next edge's weight row (global_prefetch_b8)
  __builtin_prefetch(wrow + 4096 + (lane << 7), 0, 1);

  float acc0 = 0.0f, acc1 = 0.0f;
  #pragma unroll 4
  for (int h = 0; h < 32; ++h) {
    float a = __shfl(a0, h, 32);
    union { unsigned int u; _Float16 f[2]; } p;
    p.u = *(const unsigned int*)(wrow + (h << 6) + (lane << 1));
    acc0 = fmaf(a, (float)p.f[0], acc0);
    acc1 = fmaf(a, (float)p.f[1], acc1);
  }
  #pragma unroll 4
  for (int h = 0; h < 32; ++h) {
    float a = __shfl(a1, h, 32);
    union { unsigned int u; _Float16 f[2]; } p;
    p.u = *(const unsigned int*)(wrow + ((h + 32) << 6) + (lane << 1));
    acc0 = fmaf(a, (float)p.f[0], acc0);
    acc1 = fmaf(a, (float)p.f[1], acc1);
  }
  atomicAdd(&agg[(size_t)d * 64 + 2 * lane],     acc0);
  atomicAdd(&agg[(size_t)d * 64 + 2 * lane + 1], acc1);
}

// GRU gate fusion: reads gi/gh (V,192) and h (V,64); writes h in place.
__global__ __launch_bounds__(256)
void gru_gate_kernel(const float* __restrict__ gi,
                     const float* __restrict__ gh,
                     float* __restrict__ h)
{
  int t = blockIdx.x * blockDim.x + threadIdx.x;
  if (t >= Vn * 64) return;
  int v = t >> 6, j = t & 63;
  const float* giv = gi + (size_t)v * 192;
  const float* ghv = gh + (size_t)v * 192;
  float ir = giv[j], iz = giv[64 + j], in_ = giv[128 + j];
  float hr = ghv[j], hz = ghv[64 + j], hn  = ghv[128 + j];
  float rg = 1.0f / (1.0f + __expf(-(ir + hr)));
  float zg = 1.0f / (1.0f + __expf(-(iz + hz)));
  float ng = tanhf(in_ + rg * hn);
  h[t] = (1.0f - zg) * ng + zg * h[t];
}

// (rows,cols) -> (cols,rows) transpose for GRU weights
__global__ __launch_bounds__(256)
void transpose_kernel(const float* __restrict__ in, float* __restrict__ out,
                      int rows, int cols)
{
  int t = blockIdx.x * blockDim.x + threadIdx.x;
  if (t >= rows * cols) return;
  int r = t / cols, c = t % cols;
  out[(size_t)c * rows + r] = in[t];
}

// convT weights (C=64,O=64,2,2) -> wk[kl][c][o]
__global__ __launch_bounds__(256)
void extract_convw_kernel(const float* __restrict__ w, float* __restrict__ wk)
{
  int t = blockIdx.x * blockDim.x + threadIdx.x;
  if (t >= 4 * 64 * 64) return;
  int kl = t >> 12, c = (t >> 6) & 63, o = t & 63;
  wk[t] = w[(size_t)c * 256 + o * 4 + kl];
}

// row maps for convT output scatter: rIn=(b,h,w)@rr -> rOut=(b,2h+k,2w+l)@2rr
__global__ __launch_bounds__(256)
void build_maps_kernel(int* __restrict__ map, int rr)
{
  int per = 6 * rr * rr;
  int t = blockIdx.x * blockDim.x + threadIdx.x;
  if (t >= 4 * per) return;
  int kl = t / per, rIn = t % per;
  int b = rIn / (rr * rr), rem = rIn % (rr * rr);
  int hh = rem / rr, ww = rem % rr;
  int k = kl >> 1, l = kl & 1, W2 = 2 * rr;
  map[t] = b * (W2 * W2) + (2 * hh + k) * W2 + (2 * ww + l);
}

// final tiny layer: out(98304,3) = relu'd D1(98304,32) @ dw2(32,3) + db2
__global__ __launch_bounds__(256)
void dec2_kernel(const float* __restrict__ D1, const float* __restrict__ dw2,
                 const float* __restrict__ db2, float* __restrict__ out)
{
  int row = blockIdx.x * blockDim.x + threadIdx.x;
  if (row >= 6 * Rres * Rres) return;
  const float* x = D1 + (size_t)row * 32;
  float o0 = db2[0], o1 = db2[1], o2 = db2[2];
  #pragma unroll 8
  for (int k = 0; k < 32; ++k) {
    float v = x[k];
    o0 = fmaf(v, dw2[k * 3 + 0], o0);
    o1 = fmaf(v, dw2[k * 3 + 1], o1);
    o2 = fmaf(v, dw2[k * 3 + 2], o2);
  }
  out[(size_t)row * 3 + 0] = o0;
  out[(size_t)row * 3 + 1] = o1;
  out[(size_t)row * 3 + 2] = o2;
}

// ---------------------------------------------------------------------------
// Host side
// ---------------------------------------------------------------------------
struct MpnnParams {
  const float *w1, *b1, *w2, *b2, *ew1, *eb1, *ew2, *eb2, *conv_b,
              *wih, *whh, *bih, *bhh;
};

static inline void launch_gemm(hipStream_t s,
                               const float* A, int lda, const float* B, int ldb,
                               void* C, int ldc, const float* bias,
                               const int* rmap, int M, int N, int K,
                               bool relu, bool halfout)
{
  dim3 g(M / GBM, N / GBN), b(256);
  if (halfout)
    gemm_wmma<false, true,  false><<<g, b, 0, s>>>(A, lda, B, ldb, C, ldc, bias, rmap, M, N, K);
  else if (rmap)
    gemm_wmma<false, false, true ><<<g, b, 0, s>>>(A, lda, B, ldb, C, ldc, bias, rmap, M, N, K);
  else if (relu)
    gemm_wmma<true,  false, false><<<g, b, 0, s>>>(A, lda, B, ldb, C, ldc, bias, rmap, M, N, K);
  else
    gemm_wmma<false, false, false><<<g, b, 0, s>>>(A, lda, B, ldb, C, ldc, bias, rmap, M, N, K);
}

static void run_mpnn(hipStream_t st, const MpnnParams& P,
                     const float* edge_feats, const int* src, const int* dst,
                     float* nfio,            // (V,64) in/out
                     float* tA, float* agg, float* gi, float* gh,
                     float* he, _Float16* we16, float* wihT, float* whhT)
{
  // node projection: relu(nf@w1+b1)@w2+b2
  launch_gemm(st, nfio, 64, P.w1, 64, tA,   64, P.b1, nullptr, Vn, 64, 64, true,  false);
  launch_gemm(st, tA,   64, P.w2, 64, nfio, 64, P.b2, nullptr, Vn, 64, 64, false, false);
  // edge network: he = relu(ef@ew1+eb1); we16 = (he@ew2+eb2) as f16
  launch_gemm(st, edge_feats, 64, P.ew1, 128, he, 128, P.eb1, nullptr, En, 128, 64,  true,  false);
  launch_gemm(st, he, 128, P.ew2, 4096, we16, 4096, P.eb2, nullptr, En, 4096, 128, false, true);
  // GRU weight transposes: (192,64) -> (64,192)
  transpose_kernel<<<(192 * 64 + 255) / 256, 256, 0, st>>>(P.wih, wihT, 192, 64);
  transpose_kernel<<<(192 * 64 + 255) / 256, 256, 0, st>>>(P.whh, whhT, 192, 64);

  const int nElemNF = Vn * 64;
  for (int step = 0; step < 3; ++step) {
    agg_init_kernel<<<(nElemNF + 255) / 256, 256, 0, st>>>(agg, P.conv_b);
    nnconv_msg_kernel<<<(En * 32) / 256, 256, 0, st>>>(nfio, we16, src, dst, agg);
    relu_kernel<<<(nElemNF + 255) / 256, 256, 0, st>>>(agg, nElemNF);
    launch_gemm(st, agg,  64, wihT, 192, gi, 192, P.bih, nullptr, Vn, 192, 64, false, false);
    launch_gemm(st, nfio, 64, whhT, 192, gh, 192, P.bhh, nullptr, Vn, 192, 64, false, false);
    gru_gate_kernel<<<(nElemNF + 255) / 256, 256, 0, st>>>(gi, gh, nfio);
  }
}

extern "C" void kernel_launch(void* const* d_in, const int* in_sizes, int n_in,
                              void* d_out, int out_size, void* d_ws, size_t ws_size,
                              hipStream_t stream)
{
  (void)in_sizes; (void)n_in; (void)out_size; (void)ws_size;

  const float* node_feats = (const float*)d_in[0];
  const float* edge_feats = (const float*)d_in[1];
  const int*   src        = (const int*)d_in[2];
  const int*   dst        = (const int*)d_in[3];

  auto mp = [&](int b) {
    MpnnParams P;
    P.w1 = (const float*)d_in[b + 0];  P.b1  = (const float*)d_in[b + 1];
    P.w2 = (const float*)d_in[b + 2];  P.b2  = (const float*)d_in[b + 3];
    P.ew1 = (const float*)d_in[b + 4]; P.eb1 = (const float*)d_in[b + 5];
    P.ew2 = (const float*)d_in[b + 6]; P.eb2 = (const float*)d_in[b + 7];
    P.conv_b = (const float*)d_in[b + 8];
    P.wih = (const float*)d_in[b + 9]; P.whh = (const float*)d_in[b + 10];
    P.bih = (const float*)d_in[b + 11]; P.bhh = (const float*)d_in[b + 12];
    return P;
  };
  MpnnParams Pin  = mp(4);
  MpnnParams Ppro = mp(17);
  const float* up1_w = (const float*)d_in[30];
  const float* up1_b = (const float*)d_in[31];
  const float* up2_w = (const float*)d_in[32];
  const float* up2_b = (const float*)d_in[33];
  const float* dw1   = (const float*)d_in[34];
  const float* db1   = (const float*)d_in[35];
  const float* dw2   = (const float*)d_in[36];
  const float* db2   = (const float*)d_in[37];

  // workspace carve
  char* p = (char*)d_ws;
  auto alloc = [&](size_t bytes) {
    void* r = (void*)p; p += (bytes + 255) & ~(size_t)255; return r;
  };
  float*     nf    = (float*)    alloc((size_t)Vn * 64 * 4);
  float*     tA    = (float*)    alloc((size_t)Vn * 64 * 4);
  float*     agg   = (float*)    alloc((size_t)Vn * 64 * 4);
  float*     gi    = (float*)    alloc((size_t)Vn * 192 * 4);
  float*     gh    = (float*)    alloc((size_t)Vn * 192 * 4);
  float*     he    = (float*)    alloc((size_t)En * 128 * 4);
  _Float16*  we16  = (_Float16*) alloc((size_t)En * 4096 * 2);   // 201 MB f16
  float*     wihT  = (float*)    alloc((size_t)64 * 192 * 4);
  float*     whhT  = (float*)    alloc((size_t)64 * 192 * 4);
  float*     U1    = (float*)    alloc((size_t)6 * 64 * 64 * 64 * 4);
  float*     U2    = (float*)    alloc((size_t)6 * 128 * 128 * 64 * 4);
  float*     D1    = (float*)    alloc((size_t)6 * 128 * 128 * 32 * 4);
  float*     wk1   = (float*)    alloc((size_t)4 * 64 * 64 * 4);
  float*     wk2   = (float*)    alloc((size_t)4 * 64 * 64 * 4);
  int*       map1  = (int*)      alloc((size_t)4 * 6144 * 4);
  int*       map2  = (int*)      alloc((size_t)4 * 24576 * 4);

  // encoder pooling: (6*R*R,64) -> nf (V,64)
  pool4x4_kernel<<<(Vn * 64 + 255) / 256, 256, 0, stream>>>(node_feats, nf);

  // GNN stack: input MPNN (nin=64) then 1 processor MPNN
  run_mpnn(stream, Pin,  edge_feats, src, dst, nf, tA, agg, gi, gh, he, we16, wihT, whhT);
  run_mpnn(stream, Ppro, edge_feats, src, dst, nf, tA, agg, gi, gh, he, we16, wihT, whhT);

  // decoderRes: two transposed-conv 2x2 stride-2 = 4 GEMMs each + row remap
  extract_convw_kernel<<<(4 * 4096 + 255) / 256, 256, 0, stream>>>(up1_w, wk1);
  extract_convw_kernel<<<(4 * 4096 + 255) / 256, 256, 0, stream>>>(up2_w, wk2);
  build_maps_kernel<<<(4 * 6144  + 255) / 256, 256, 0, stream>>>(map1, 32);
  build_maps_kernel<<<(4 * 24576 + 255) / 256, 256, 0, stream>>>(map2, 64);

  for (int kl = 0; kl < 4; ++kl)
    launch_gemm(stream, nf, 64, wk1 + kl * 4096, 64, U1, 64, up1_b,
                map1 + kl * 6144, Vn, 64, 64, false, false);
  for (int kl = 0; kl < 4; ++kl)
    launch_gemm(stream, U1, 64, wk2 + kl * 4096, 64, U2, 64, up2_b,
                map2 + kl * 24576, 6 * 64 * 64, 64, 64, false, false);

  // decoder MLP: relu(U2@dw1+db1) via WMMA, then tiny N=3 layer
  launch_gemm(stream, U2, 64, dw1, 32, D1, 32, db1, nullptr,
              6 * Rres * Rres, 32, 64, true, false);
  dec2_kernel<<<(6 * Rres * Rres + 255) / 256, 256, 0, stream>>>(
      D1, dw2, db2, (float*)d_out);
}
